// TopKRouter_52544629899282
// MI455X (gfx1250) — matrix operand order repaired and checked by reading, not compiled, not run on hardware
//
#include <hip/hip_runtime.h>

// Problem constants (from reference): B=4, S=4096, D=4096, E=64, K=2
#define DD 4096
#define EE 64
#define NROWS 16384        // B*S
#define ROWS_PER_BLOCK 64
#define NBLOCKS (NROWS / ROWS_PER_BLOCK)   // 256

// Output layout (floats, concatenated in reference return order)
#define OFF_IDX    0            // [B,S,K] = 32768
#define OFF_SCORES 32768        // 32768
#define OFF_PROBS  65536        // B*S*E = 1048576
#define OFF_Z      1114112      // scalar
#define OFF_IMP    1114113      // 64
#define OFF_LOAD   1114177      // 64  -> total 1114241

typedef __attribute__((ext_vector_type(16))) __bf16 v16bf;
typedef __attribute__((ext_vector_type(8)))  float  v8f;

// ---------------------------------------------------------------- init: zero accumulators
__global__ void router_init_kernel(float* __restrict__ z_imp, float* __restrict__ cnt) {
    int t = threadIdx.x;
    if (t < 65) z_imp[t] = 0.0f;   // z at [0], importance sums at [1..64]
    if (t < 64) cnt[t]   = 0.0f;
}

// ---------------------------------------------------------------- W fp32 -> bf16 (once per call)
__global__ void wgate_cvt_kernel(const float* __restrict__ W, __bf16* __restrict__ wbf, int n) {
    int i = blockIdx.x * blockDim.x + threadIdx.x;
    if (i < n) wbf[i] = (__bf16)W[i];
}

// ---------------------------------------------------------------- main: WMMA gemm + softmax/top2/stats
__global__ __launch_bounds__(256) void router_main_kernel(
    const float* __restrict__ x, const __bf16* __restrict__ wbf,
    float* __restrict__ out_idx, float* __restrict__ out_scores,
    float* __restrict__ out_probs, float* __restrict__ out_z,
    float* __restrict__ out_imp, float* __restrict__ cnt_ws)
{
    __shared__ float tileA[ROWS_PER_BLOCK * EE];   // k-half 0 partial logits / later exp values
    __shared__ float tileB[ROWS_PER_BLOCK * EE];   // k-half 1 partial logits
    __shared__ float rowInv[ROWS_PER_BLOCK];
    __shared__ float ldsCnt[EE];
    __shared__ float ldsZ;

    const int tid   = threadIdx.x;
    const int wave  = tid >> 5;          // wave32
    const int lane  = tid & 31;
    const int waveM = wave & 3;          // which 16-row group
    const int kHalf = wave >> 2;         // 0: K [0,2048)  1: K [2048,4096)
    const int blockRow = blockIdx.x * ROWS_PER_BLOCK;

    if (tid < EE) ldsCnt[tid] = 0.0f;
    if (tid == 0) ldsZ = 0.0f;

    // ---- per-lane fragment addressing (contiguous k = 16*half + s, same for A and B) ----
    const int half  = lane >> 4;         // 0 or 1
    const int ln    = lane & 15;
    const int aRow  = blockRow + waveM * 16 + ln;   // A: M = ln
    const int aKoff = half * 16;         // A: lanes 16-31 hold K 16..31
    const int bKoff = half * 16;         // B: lanes 16-31 hold K 16..31
    const float* xrow = x + (size_t)aRow * DD;
    const __bf16* wp0 = wbf + (size_t)( 0 + ln) * DD + bKoff;
    const __bf16* wp1 = wbf + (size_t)(16 + ln) * DD + bKoff;
    const __bf16* wp2 = wbf + (size_t)(32 + ln) * DD + bKoff;
    const __bf16* wp3 = wbf + (size_t)(48 + ln) * DD + bKoff;

    v8f acc0 = {}, acc1 = {}, acc2 = {}, acc3 = {};

    const int kBeg = kHalf * (DD / 2);
    const int kEnd = kBeg + (DD / 2);
    for (int kb = kBeg; kb < kEnd; kb += 32) {
        __builtin_prefetch(xrow + kb + 64, 0, 0);   // global_prefetch_b8, next x chunk
        const float4 a0 = *(const float4*)(xrow + kb + aKoff);
        const float4 a1 = *(const float4*)(xrow + kb + aKoff + 4);
        const float4 a2 = *(const float4*)(xrow + kb + aKoff + 8);
        const float4 a3 = *(const float4*)(xrow + kb + aKoff + 12);
        v16bf A;
        A[0]  = (__bf16)a0.x; A[1]  = (__bf16)a0.y; A[2]  = (__bf16)a0.z; A[3]  = (__bf16)a0.w;
        A[4]  = (__bf16)a1.x; A[5]  = (__bf16)a1.y; A[6]  = (__bf16)a1.z; A[7]  = (__bf16)a1.w;
        A[8]  = (__bf16)a2.x; A[9]  = (__bf16)a2.y; A[10] = (__bf16)a2.z; A[11] = (__bf16)a2.w;
        A[12] = (__bf16)a3.x; A[13] = (__bf16)a3.y; A[14] = (__bf16)a3.z; A[15] = (__bf16)a3.w;

        const v16bf B0 = *(const v16bf*)(wp0 + kb);
        const v16bf B1 = *(const v16bf*)(wp1 + kb);
        const v16bf B2 = *(const v16bf*)(wp2 + kb);
        const v16bf B3 = *(const v16bf*)(wp3 + kb);

        acc0 = __builtin_amdgcn_wmma_f32_16x16x32_bf16(false, A, false, B0, (short)0, acc0, false, false);
        acc1 = __builtin_amdgcn_wmma_f32_16x16x32_bf16(false, A, false, B1, (short)0, acc1, false, false);
        acc2 = __builtin_amdgcn_wmma_f32_16x16x32_bf16(false, A, false, B2, (short)0, acc2, false, false);
        acc3 = __builtin_amdgcn_wmma_f32_16x16x32_bf16(false, A, false, B3, (short)0, acc3, false, false);
    }

    // ---- dump 16x64 partial-logit tile to LDS (C/D layout: lane n=ln, M = r + 8*half) ----
    {
        float* tile = kHalf ? tileB : tileA;
        float* tr = tile + (waveM * 16 + half * 8) * EE + ln;
        #pragma unroll
        for (int r = 0; r < 8; ++r) {
            tr[r * EE +  0] = acc0[r];
            tr[r * EE + 16] = acc1[r];
            tr[r * EE + 32] = acc2[r];
            tr[r * EE + 48] = acc3[r];
        }
    }
    __syncthreads();

    // ---- per-row: max, top-2, exp/sum, z^2, counts, idx/scores ----
    if (tid < ROWS_PER_BLOCK) {
        const int r = tid;
        float m1 = -3.402823466e38f, m2 = -3.402823466e38f;
        int i1 = 0, i2 = 0;
        for (int e = 0; e < EE; ++e) {
            float v = tileA[r * EE + e] + tileB[r * EE + e];
            if (v > m1)      { m2 = m1; i2 = i1; m1 = v; i1 = e; }
            else if (v > m2) { m2 = v;  i2 = e; }
        }
        float s = 0.0f;
        for (int e = 0; e < EE; ++e) {
            float v  = tileA[r * EE + e] + tileB[r * EE + e];
            float ex = __expf(v - m1);
            tileA[r * EE + e] = ex;           // keep exp values for probs/importance
            s += ex;
        }
        rowInv[r] = 1.0f / s;
        float z = m1 + __logf(s);
        atomicAdd(&ldsZ, z * z);
        atomicAdd(&ldsCnt[i1], 1.0f);
        atomicAdd(&ldsCnt[i2], 1.0f);
        float d21 = __expf(m2 - m1);          // softmax over [m1, m2] (descending, as top_k)
        float s1  = 1.0f / (1.0f + d21);
        size_t row = (size_t)blockRow + r;
        out_idx[row * 2 + 0]    = (float)i1;
        out_idx[row * 2 + 1]    = (float)i2;
        out_scores[row * 2 + 0] = s1;
        out_scores[row * 2 + 1] = 1.0f - s1;
    }
    __syncthreads();

    // ---- coalesced probs write: 64 rows * 64 experts = 4096 elems, 16 per thread ----
    {
        float* pb = out_probs + (size_t)blockRow * EE;
        #pragma unroll
        for (int c = 0; c < 16; ++c) {
            int i = c * 256 + tid;
            pb[i] = tileA[i] * rowInv[i >> 6];
        }
    }
    // ---- per-expert block partials -> global atomics ----
    if (tid < EE) {
        float sum = 0.0f;
        for (int r = 0; r < ROWS_PER_BLOCK; ++r)
            sum += tileA[r * EE + tid] * rowInv[r];
        atomicAdd(&out_imp[tid], sum);
        atomicAdd(&cnt_ws[tid], ldsCnt[tid]);
    }
    if (tid == 0) atomicAdd(out_z, ldsZ);
}

// ---------------------------------------------------------------- finalize: scale sums
__global__ void router_fin_kernel(float* __restrict__ z, float* __restrict__ imp,
                                  float* __restrict__ loadOut, const float* __restrict__ cnt) {
    int t = threadIdx.x;
    if (t == 0) z[0] *= (1.0f / 16384.0f);                    // mean over B*S rows
    if (t < 64) {
        imp[t]     *= (1.0f / 16384.0f);                      // mean probs
        loadOut[t]  = cnt[t] * (1.0f / 32768.0f);             // counts / (B*S*K)
    }
}

extern "C" void kernel_launch(void* const* d_in, const int* in_sizes, int n_in,
                              void* d_out, int out_size, void* d_ws, size_t ws_size,
                              hipStream_t stream) {
    const float* x = (const float*)d_in[0];        // [4,4096,4096] fp32
    const float* W = (const float*)d_in[1];        // [64,4096] fp32

    float* out        = (float*)d_out;
    float* out_idx    = out + OFF_IDX;
    float* out_scores = out + OFF_SCORES;
    float* out_probs  = out + OFF_PROBS;
    float* out_z      = out + OFF_Z;
    float* out_imp    = out + OFF_IMP;
    float* out_load   = out + OFF_LOAD;

    float*  cnt = (float*)d_ws;                       // 64 floats
    __bf16* wbf = (__bf16*)((char*)d_ws + 512);       // 64*4096 bf16 = 512 KB

    router_init_kernel<<<1, 128, 0, stream>>>(out_z, cnt);
    wgate_cvt_kernel<<<(EE * DD + 255) / 256, 256, 0, stream>>>(W, wbf, EE * DD);
    router_main_kernel<<<NBLOCKS, 256, 0, stream>>>(x, wbf, out_idx, out_scores,
                                                    out_probs, out_z, out_imp, cnt);
    router_fin_kernel<<<1, 64, 0, stream>>>(out_z, out_imp, out_load, cnt);
}
